// UncertaintyGuidedAttention_90761248899622
// MI455X (gfx1250) — matrix operand order repaired and checked by reading, not compiled, not run on hardware
//
#include <hip/hip_runtime.h>

// ---------------------------------------------------------------------------
// Problem constants
// ---------------------------------------------------------------------------
#define BATCH 64
#define NTOK  197
#define NPAD  224            // padded token count (multiple of 32 for WMMA K)
#define DIM   768
#define NH    12
#define HD    64
#define D3    2304           // 3*DIM
#define NC    14
#define MROWS (BATCH * NTOK) // 12608 = 788*16 = 394*32

typedef __attribute__((ext_vector_type(16))) __bf16 v16bf;
typedef __attribute__((ext_vector_type(8)))  float  v8f;

// float -> bf16, round to nearest even
__device__ __forceinline__ unsigned short f2bf(float f) {
  unsigned int u = __builtin_bit_cast(unsigned int, f);
  u += 0x7FFFu + ((u >> 16) & 1u);
  return (unsigned short)(u >> 16);
}

union FragU { uint4 u[2]; v16bf v; };

// A-fragment (16-bit, 16x32): lane m = lane&15, half = lane>>4
// element e holds K = (e&7) + (e>>3)*16 + half*8 ; base points at row + k0
__device__ __forceinline__ v16bf frag_a(const unsigned short* p, int hf) {
  FragU f;
  f.u[0] = *(const uint4*)(p + hf * 8);
  f.u[1] = *(const uint4*)(p + 16 + hf * 8);
  return f.v;
}

// B-fragment (16-bit, 32x16): lane n = lane&15, half = lane>>4
// element e holds K = (e&7) + (e>>3)*16 + half*8 ; base points at (column-major row) + k0
__device__ __forceinline__ v16bf frag_b(const unsigned short* p, int hf) {
  FragU f;
  f.u[0] = *(const uint4*)(p + hf * 8);
  f.u[1] = *(const uint4*)(p + 16 + hf * 8);
  return f.v;
}

__device__ __forceinline__ v8f wmma_bf16(v16bf a, v16bf b, v8f c) {
  return __builtin_amdgcn_wmma_f32_16x16x32_bf16(false, a, false, b,
                                                 (short)0, c, false, false);
}

// ---------------------------------------------------------------------------
// Workspace layout (bytes, all 256-aligned naturally)
// ---------------------------------------------------------------------------
static constexpr size_t QB_BYTES = (size_t)BATCH * NH * NPAD * HD * 2; // 22,020,096
static constexpr size_t PB_BYTES = (size_t)BATCH * NTOK * NTOK * 4;    //  9,935,104
static constexpr size_t XB_BYTES = (size_t)MROWS * DIM * 2;            // 19,365,888
static constexpr size_t WQ_BYTES = (size_t)D3 * DIM * 2;               //  3,538,944
static constexpr size_t WP_BYTES = (size_t)DIM * DIM * 2;              //  1,179,648
static constexpr size_t AO_BYTES = (size_t)MROWS * DIM * 2;            // 19,365,888

static constexpr size_t O_QB   = 0;
static constexpr size_t O_KB   = O_QB + QB_BYTES;
static constexpr size_t O_VT   = O_KB + QB_BYTES;
static constexpr size_t O_PB   = O_VT + QB_BYTES;          // zero region ends after pbuf
static constexpr size_t O_XB   = O_PB + PB_BYTES;
static constexpr size_t O_WQ   = O_XB + XB_BYTES;
static constexpr size_t O_WP   = O_WQ + WQ_BYTES;
static constexpr size_t O_AO   = O_WP + WP_BYTES;
static constexpr size_t O_COEF = O_AO + AO_BYTES;
static constexpr size_t WS_NEEDED = O_COEF + 4096;
static constexpr size_t ZERO_U4 = (3 * QB_BYTES + PB_BYTES) / 16;      // 4,749,712

// ---------------------------------------------------------------------------
__global__ void k_zero(uint4* __restrict__ p, size_t n) {
  size_t i = (size_t)blockIdx.x * blockDim.x + threadIdx.x;
  const size_t st = (size_t)gridDim.x * blockDim.x;
  const uint4 z = make_uint4(0u, 0u, 0u, 0u);
  for (; i < n; i += st) p[i] = z;
}

// 8 floats -> 8 bf16 per thread (two b128 loads, one b128 store)
__global__ void k_cvt8(const float* __restrict__ src, unsigned short* __restrict__ dst,
                       size_t n8) {
  size_t i = (size_t)blockIdx.x * blockDim.x + threadIdx.x;
  const size_t st = (size_t)gridDim.x * blockDim.x;
  for (; i < n8; i += st) {
    const float4 f0 = ((const float4*)src)[2 * i];
    const float4 f1 = ((const float4*)src)[2 * i + 1];
    union { unsigned short s[8]; uint4 u; } o;
    o.s[0] = f2bf(f0.x); o.s[1] = f2bf(f0.y); o.s[2] = f2bf(f0.z); o.s[3] = f2bf(f0.w);
    o.s[4] = f2bf(f1.x); o.s[5] = f2bf(f1.y); o.s[6] = f2bf(f1.z); o.s[7] = f2bf(f1.w);
    ((uint4*)dst)[i] = o.u;
  }
}

// dst[n*K + k] = bf16(src[k*Nc + n])  (store weights column-major as bf16)
__global__ void k_cvt_t(const float* __restrict__ src, unsigned short* __restrict__ dst,
                        int K, int Nc) {
  size_t i = (size_t)blockIdx.x * blockDim.x + threadIdx.x;
  const size_t st = (size_t)gridDim.x * blockDim.x;
  const size_t total = (size_t)K * Nc;
  for (; i < total; i += st) {
    const int n = (int)(i / K);
    const int k = (int)(i - (size_t)n * K);
    dst[i] = f2bf(src[(size_t)k * Nc + n]);
  }
}

// uncertainty MLP -> fused per-(b,h) score coefficient
__global__ void k_mlp(const float* __restrict__ u,
                      const float* __restrict__ W1, const float* __restrict__ b1,
                      const float* __restrict__ W2, const float* __restrict__ b2,
                      const float* __restrict__ W3, const float* __restrict__ b3,
                      const int* __restrict__ uga_p, float* __restrict__ coef) {
  const int b = threadIdx.x;
  if (b >= BATCH) return;
  float uu[NC], um = 0.f;
#pragma unroll
  for (int i = 0; i < NC; ++i) { uu[i] = u[b * NC + i]; um += uu[i]; }
  um *= (1.0f / NC);
  float h1[64];
  for (int j = 0; j < 64; ++j) {
    float s = b1[j];
#pragma unroll
    for (int i = 0; i < NC; ++i) s += uu[i] * W1[i * 64 + j];
    h1[j] = fmaxf(s, 0.f);
  }
  float h2[128];
  for (int j = 0; j < 128; ++j) {
    float s = b2[j];
    for (int i = 0; i < 64; ++i) s += h1[i] * W2[i * 128 + j];
    h2[j] = fmaxf(s, 0.f);
  }
  const float uga = (float)(*uga_p);
  const float scale = 0.125f;  // 64^-0.5
  for (int hh = 0; hh < NH; ++hh) {
    float s = b3[hh];
    for (int i = 0; i < 128; ++i) s += h2[i] * W3[i * NH + hh];
    const float mod = 1.0f / (1.0f + __expf(-s));
    float c = scale;
    if (uga > 0.f)
      c = scale * (1.0f + uga * mod) / (8.0f * (1.0f + 0.5f * uga * um));
    coef[b * NH + hh] = c;
  }
}

// QKV GEMM, 2x4 register-blocked: each wave computes a 32x64 C patch
// (12608x768) bf16 @ (768x2304) bf16 -> scatter q/k/v (v transposed)
__global__ void __launch_bounds__(128)
k_qkv(const unsigned short* __restrict__ xb, const unsigned short* __restrict__ wT,
      const float* __restrict__ bqkv, unsigned short* __restrict__ qb,
      unsigned short* __restrict__ kb, unsigned short* __restrict__ vtb) {
  const int wave = threadIdx.x >> 5;
  const int lane = threadIdx.x & 31;
  const int ls = lane & 15;
  const int hf = lane >> 4;
  const int m0 = blockIdx.x * 32;
  const int n0 = (blockIdx.y * 4 + wave) * 64;
  const unsigned short* arow0 = xb + (size_t)(m0 + ls) * DIM;
  const unsigned short* arow1 = arow0 + (size_t)16 * DIM;
  const unsigned short* brow0 = wT + (size_t)(n0 + ls) * DIM;
  v8f acc[2][4] = {};
#pragma unroll 2
  for (int k0 = 0; k0 < DIM; k0 += 32) {
    const v16bf a0 = frag_a(arow0 + k0, hf);
    const v16bf a1 = frag_a(arow1 + k0, hf);
#pragma unroll
    for (int j = 0; j < 4; ++j) {
      const v16bf bb = frag_b(brow0 + (size_t)j * 16 * DIM + k0, hf);
      acc[0][j] = wmma_bf16(a0, bb, acc[0][j]);
      acc[1][j] = wmma_bf16(a1, bb, acc[1][j]);
    }
  }
#pragma unroll
  for (int j = 0; j < 4; ++j) {
    const int col = n0 + j * 16 + ls;
    const float bias = bqkv[col];
    const int which = col / DIM;
    const int h  = (col >> 6) % NH;
    const int hd = col & 63;
    const size_t voff = (size_t)hd * NPAD;
#pragma unroll
    for (int i = 0; i < 2; ++i) {
#pragma unroll
      for (int r = 0; r < 8; ++r) {
        const int row = m0 + i * 16 + r + hf * 8;
        const int b = row / NTOK;
        const int n = row - b * NTOK;
        const unsigned short val = f2bf(acc[i][j][r] + bias);
        const size_t bh = (size_t)(b * NH + h);
        if (which == 0)      qb[(bh * NPAD + n) * HD + hd] = val;
        else if (which == 1) kb[(bh * NPAD + n) * HD + hd] = val;
        else                 vtb[bh * (HD * NPAD) + voff + n] = val; // transposed V
      }
    }
  }
}

// Fused attention tile: S = c*(QK^T), exact softmax, entropy accumulate, P@V
__global__ void __launch_bounds__(128)
k_attn(const unsigned short* __restrict__ qb, const unsigned short* __restrict__ kb,
       const unsigned short* __restrict__ vtb, const float* __restrict__ coef,
       float* __restrict__ pbuf, unsigned short* __restrict__ attn_out) {
  __shared__ __align__(16) float S[16][NPAD];
  __shared__ __align__(16) unsigned short Pb[16][NPAD];
  const int mt = blockIdx.x, h = blockIdx.y, b = blockIdx.z;
  const int wave = threadIdx.x >> 5;
  const int lane = threadIdx.x & 31;
  const int ls = lane & 15;
  const int hf = lane >> 4;
  const size_t bh = (size_t)(b * NH + h);
  const float cf = coef[bh];

  // ---- phase 1: S tiles via WMMA (A = Q rows, B = K rows as K^T columns) ----
  const unsigned short* qrow = qb + (bh * NPAD + mt * 16 + ls) * HD;
  const v16bf a0 = frag_a(qrow, hf);
  const v16bf a1 = frag_a(qrow + 32, hf);
  for (int ct = wave; ct < 13; ct += 4) {
    const unsigned short* krow = kb + (bh * NPAD + ct * 16 + ls) * HD;
    v8f acc = {};
    acc = wmma_bf16(a0, frag_b(krow, hf), acc);
    acc = wmma_bf16(a1, frag_b(krow + 32, hf), acc);
#pragma unroll
    for (int r = 0; r < 8; ++r)
      S[r + hf * 8][ct * 16 + ls] = acc[r] * cf;
  }
  // warm L1/L2 for phase-3 V rows while softmax runs (global_prefetch_b8)
  const unsigned short* vrow = vtb + (bh * HD + wave * 16 + ls) * NPAD;
  __builtin_prefetch(vrow, 0, 3);
  __builtin_prefetch(vrow + 64, 0, 3);
  __builtin_prefetch(vrow + 128, 0, 3);
  __builtin_prefetch(vrow + 192, 0, 3);
  __syncthreads();

  // ---- phase 2: cooperative row softmax (8 lanes/row) + entropy accumulate ----
  {
    const int row = threadIdx.x >> 3;   // 0..15
    const int sub = threadIdx.x & 7;    // 0..7
    const int ng = mt * 16 + row;
    const bool live = (ng < NTOK);
    float mx = -3.0e38f;
    if (live)
      for (int m = sub; m < NTOK; m += 8) mx = fmaxf(mx, S[row][m]);
#pragma unroll
    for (int o = 4; o; o >>= 1) mx = fmaxf(mx, __shfl_xor(mx, o, 8));
    float sum = 0.f;
    if (live)
      for (int m = sub; m < NTOK; m += 8) {
        const float e = __expf(S[row][m] - mx);
        S[row][m] = e; sum += e;
      }
#pragma unroll
    for (int o = 4; o; o >>= 1) sum += __shfl_xor(sum, o, 8);
    if (live) {
      const float inv  = 1.0f / sum;
      const float invH = inv * (1.0f / NH);
      float* prow = pbuf + ((size_t)b * NTOK + ng) * NTOK;
      for (int m = sub; m < NTOK; m += 8) {
        const float e = S[row][m];
        Pb[row][m] = f2bf(e * inv);
        atomicAdd(prow + m, e * invH);
      }
      for (int m = NTOK + sub; m < NPAD; m += 8) Pb[row][m] = 0;
    } else {
      for (int m = sub; m < NPAD; m += 8) Pb[row][m] = 0;
    }
  }
  __syncthreads();

  // ---- phase 3: out tile = P (16xNPAD) @ V (NPADx64), one 16-col tile/wave ----
  v8f acc = {};
#pragma unroll
  for (int k0 = 0; k0 < NPAD; k0 += 32) {
    const v16bf pa = frag_a((const unsigned short*)&Pb[ls][0] + k0, hf);
    acc = wmma_bf16(pa, frag_b(vrow + k0, hf), acc);
  }
  const int n0 = wave * 16;
#pragma unroll
  for (int r = 0; r < 8; ++r) {
    const int ng = mt * 16 + r + hf * 8;
    if (ng < NTOK)
      attn_out[(size_t)(b * NTOK + ng) * DIM + h * HD + n0 + ls] = f2bf(acc[r]);
  }
}

// entropy of head-averaged attention, per batch; tail of d_out
__global__ void k_ent(const float* __restrict__ pbuf, float* __restrict__ out_tail) {
  __shared__ float red[256];
  const int b = blockIdx.x;
  const float* p = pbuf + (size_t)b * (NTOK * NTOK);
  float s = 0.f;
  for (int i = threadIdx.x; i < NTOK * NTOK; i += 256) {
    const float v = p[i] + 1e-8f;
    s += v * __logf(v);
  }
  red[threadIdx.x] = s;
  __syncthreads();
  for (int st = 128; st > 0; st >>= 1) {
    if (threadIdx.x < st) red[threadIdx.x] += red[threadIdx.x + st];
    __syncthreads();
  }
  if (threadIdx.x == 0) out_tail[b] = -red[0] * (1.0f / NTOK);
  if (b == 0 && threadIdx.x == 0) out_tail[BATCH] = __logf((float)NTOK);
}

// Projection GEMM, 2x4 register-blocked: attn_out bf16 @ WprojT + bproj -> f32
__global__ void __launch_bounds__(128)
k_proj(const unsigned short* __restrict__ ab, const unsigned short* __restrict__ wT,
       const float* __restrict__ bproj, float* __restrict__ out) {
  const int wave = threadIdx.x >> 5;
  const int lane = threadIdx.x & 31;
  const int ls = lane & 15;
  const int hf = lane >> 4;
  const int m0 = blockIdx.x * 32;
  const int n0 = (blockIdx.y * 4 + wave) * 64;
  const unsigned short* arow0 = ab + (size_t)(m0 + ls) * DIM;
  const unsigned short* arow1 = arow0 + (size_t)16 * DIM;
  const unsigned short* brow0 = wT + (size_t)(n0 + ls) * DIM;
  v8f acc[2][4] = {};
#pragma unroll 2
  for (int k0 = 0; k0 < DIM; k0 += 32) {
    const v16bf a0 = frag_a(arow0 + k0, hf);
    const v16bf a1 = frag_a(arow1 + k0, hf);
#pragma unroll
    for (int j = 0; j < 4; ++j) {
      const v16bf bb = frag_b(brow0 + (size_t)j * 16 * DIM + k0, hf);
      acc[0][j] = wmma_bf16(a0, bb, acc[0][j]);
      acc[1][j] = wmma_bf16(a1, bb, acc[1][j]);
    }
  }
#pragma unroll
  for (int j = 0; j < 4; ++j) {
    const int col = n0 + j * 16 + ls;
    const float bias = bproj[col];
#pragma unroll
    for (int i = 0; i < 2; ++i) {
#pragma unroll
      for (int r = 0; r < 8; ++r) {
        const int row = m0 + i * 16 + r + hf * 8;
        out[(size_t)row * DIM + col] = acc[i][j][r] + bias;
      }
    }
  }
}

// ---------------------------------------------------------------------------
extern "C" void kernel_launch(void* const* d_in, const int* in_sizes, int n_in,
                              void* d_out, int out_size, void* d_ws, size_t ws_size,
                              hipStream_t stream) {
  (void)in_sizes; (void)n_in; (void)out_size;
  if (ws_size < WS_NEEDED) return;

  const float* x     = (const float*)d_in[0];
  const float* u     = (const float*)d_in[1];
  const float* Wqkv  = (const float*)d_in[2];
  const float* bqkv  = (const float*)d_in[3];
  const float* Wproj = (const float*)d_in[4];
  const float* bproj = (const float*)d_in[5];
  const float* W1 = (const float*)d_in[6];
  const float* b1 = (const float*)d_in[7];
  const float* W2 = (const float*)d_in[8];
  const float* b2 = (const float*)d_in[9];
  const float* W3 = (const float*)d_in[10];
  const float* b3 = (const float*)d_in[11];
  const int* uga  = (const int*)d_in[12];
  float* out = (float*)d_out;

  char* ws = (char*)d_ws;
  unsigned short* qb   = (unsigned short*)(ws + O_QB);
  unsigned short* kb   = (unsigned short*)(ws + O_KB);
  unsigned short* vtb  = (unsigned short*)(ws + O_VT);
  float*          pbuf = (float*)(ws + O_PB);
  unsigned short* xb   = (unsigned short*)(ws + O_XB);
  unsigned short* wqT  = (unsigned short*)(ws + O_WQ);
  unsigned short* wpT  = (unsigned short*)(ws + O_WP);
  unsigned short* ao   = (unsigned short*)(ws + O_AO);
  float*          coef = (float*)(ws + O_COEF);

  k_zero<<<2048, 256, 0, stream>>>((uint4*)ws, ZERO_U4);
  k_cvt8<<<2048, 256, 0, stream>>>(x, xb, (size_t)MROWS * DIM / 8);
  k_cvt_t<<<2048, 256, 0, stream>>>(Wqkv, wqT, DIM, D3);
  k_cvt_t<<<1024, 256, 0, stream>>>(Wproj, wpT, DIM, DIM);
  k_mlp<<<1, 64, 0, stream>>>(u, W1, b1, W2, b2, W3, b3, uga, coef);
  k_qkv<<<dim3(MROWS / 32, D3 / 256), 128, 0, stream>>>(xb, wqT, bqkv, qb, kb, vtb);
  k_attn<<<dim3(13, NH, BATCH), 128, 0, stream>>>(qb, kb, vtb, coef, pbuf, ao);
  k_ent<<<BATCH, 256, 0, stream>>>(pbuf, out + (size_t)MROWS * DIM);
  k_proj<<<dim3(MROWS / 32, DIM / 256), 128, 0, stream>>>(ao, wpT, bproj, out);
}
